// HybridQuanvolutionNAT_65481071402882
// MI455X (gfx1250) — compile-verified, matrix operand order checked by reading, not run on hardware
//
#include <hip/hip_runtime.h>
#include <hip/hip_bf16.h>
#include <math.h>

typedef __attribute__((ext_vector_type(16))) _Float16 v16h;
typedef __attribute__((ext_vector_type(8)))  float    v8f;

#define PADW  58
#define PLANE (58 * 58)   // 3364 halves per channel plane (with halo)

// ---------------------------------------------------------------------------
// Kernel 1: conv1 (1->8, 3x3, SAME) + ReLU + 2x2 maxpool.  K=9, 1 input
// channel: too skinny for the matrix pipe, memory-bound -> plain VALU.
// One thread per pooled output element.  h: [512, 8, 56, 56]
// ---------------------------------------------------------------------------
__global__ void conv1_pool_kernel(const float* __restrict__ x,
                                  const float* __restrict__ w1,
                                  const float* __restrict__ b1,
                                  float* __restrict__ h) {
  const int t  = blockIdx.x * blockDim.x + threadIdx.x;   // < 512*8*56*56
  const int px = t % 56;
  const int py = (t / 56) % 56;
  const int co = (t / 3136) % 8;
  const int b  = t / 25088;

  float w[9];
#pragma unroll
  for (int i = 0; i < 9; ++i) w[i] = w1[co * 9 + i];
  const float bias = b1[co];
  const float* xb = x + b * 112 * 112;

  float smax = -1e30f;
#pragma unroll
  for (int dy = 0; dy < 2; ++dy) {
#pragma unroll
    for (int dx = 0; dx < 2; ++dx) {
      const int y0 = 2 * py + dy;
      const int x0 = 2 * px + dx;
      float s = bias;
#pragma unroll
      for (int ky = 0; ky < 3; ++ky) {
#pragma unroll
        for (int kx = 0; kx < 3; ++kx) {
          const int yy = y0 + ky - 1;
          const int xx = x0 + kx - 1;
          float v = (yy >= 0 && yy < 112 && xx >= 0 && xx < 112)
                        ? xb[yy * 112 + xx] : 0.0f;
          s = fmaf(v, w[ky * 3 + kx], s);
        }
      }
      smax = fmaxf(smax, s);
    }
  }
  h[t] = fmaxf(smax, 0.0f);   // max(relu(s_i)) == relu(max(s_i))
}

// ---------------------------------------------------------------------------
// Kernel 2: conv2 (8->16, 3x3, SAME) + ReLU + 2x2 maxpool as LDS-staged
// im2col WMMA.  One block = one image: h[b] staged to LDS as f16 with a
// 1-px zero halo ([8][58][58], 52.6 KB).  im2col K-order is OURS (must only
// match between A and B): k = tap*8 + c, so c = jv&7 (compile-time) and
// tap = t*4 + 2*(jv>>3) + half -> all LDS/weight offsets fold to constants
// selected by the lane's half bit.  B fragments are quad-invariant and
// built once per wave.  M rows = pooled-quad * 2x2, so ReLU+maxpool fuses
// straight out of the C fragment.  f: [512, 16, 28, 28]
// ---------------------------------------------------------------------------
__global__ void __launch_bounds__(128)
conv2_wmma_pool_kernel(const float* __restrict__ h,
                       const float* __restrict__ w2,
                       const float* __restrict__ b2,
                       float* __restrict__ f) {
  __shared__ _Float16 sh[8 * PLANE];    // padded f16 image, 53,824 B

  const int tid  = threadIdx.x;         // 0..127 (4 waves)
  const int b    = blockIdx.x;          // image index
  const int lane = tid & 31;
  const int wv   = tid >> 5;            // wave in block 0..3
  const int hi   = lane >> 4;           // K-half select (ISA 7.12.2 layout)
  const int lcol = lane & 15;           // A: M row ; B/D: N column

  // ---- stage h[b] -> LDS f16 with zero halo ----
  {
    unsigned* shw = (unsigned*)sh;
    for (int i = tid; i < (8 * PLANE) / 2; i += 128) shw[i] = 0u;
    __syncthreads();                    // zeros visible before interior fill
    const float* hb = h + b * 25088;
#pragma unroll 1
    for (int i = tid; i < 25088; i += 128) {
      __builtin_prefetch(hb + i + 2048, 0, 1);   // global_prefetch_b8
      const int c  = i / 3136;
      const int r2 = i - c * 3136;
      const int yy = r2 / 56;
      const int xx = r2 - yy * 56;
      sh[c * PLANE + (yy + 1) * PADW + (xx + 1)] = (_Float16)hb[i];
    }
    __syncthreads();
  }

  // ---- build B fragments once per wave (weights are quad-invariant) ----
  // B[k][n] = w2[n*72 + c*9 + tap]  (OIHW), n = lcol
  v16h bf[3];
  const float* wn = w2 + lcol * 72;
#pragma unroll
  for (int t = 0; t < 3; ++t) {
#pragma unroll
    for (int jv = 0; jv < 16; ++jv) {
      const int c    = jv & 7;                   // compile-time
      const int tap0 = t * 4 + ((jv >> 3) << 1); // compile-time
      const int tap1 = tap0 + 1;
      float bwv;
      if (tap1 < 9)      bwv = wn[c * 9 + (hi ? tap1 : tap0)];
      else if (tap0 < 9) { float v0 = wn[c * 9 + tap0]; bwv = hi ? 0.0f : v0; }
      else               bwv = 0.0f;
      bf[t][jv] = (_Float16)bwv;
    }
  }

  // ---- 49 pooled-quads per wave ----
  const int jq  = lcol >> 2;            // pooled pixel within quad
  const int sub = lcol & 3;             // 2x2 sub-position
  const int dy  = sub >> 1, dx = sub & 1;
  const float bias = b2[lcol];

#pragma unroll 1
  for (int q = wv; q < 196; q += 4) {
    const int py  = q / 7;                       // 7 quads per pooled row
    const int px0 = (q - py * 7) * 4;
    const int y   = 2 * py + dy;                 // pre-pool row   0..55
    const int x   = 2 * (px0 + jq) + dx;         // pre-pool col   0..55
    const _Float16* lb = &sh[y * PADW + x];      // halo absorbs the -1

    v8f acc = {};
#pragma unroll
    for (int t = 0; t < 3; ++t) {
      v16h av;
#pragma unroll
      for (int jv = 0; jv < 16; ++jv) {
        const int c    = jv & 7;
        const int tap0 = t * 4 + ((jv >> 3) << 1);
        const int tap1 = tap0 + 1;
        const int off0 = c * PLANE + (tap0 / 3) * PADW + (tap0 % 3);
        const int off1 = c * PLANE + (tap1 / 3) * PADW + (tap1 % 3);
        _Float16 a;
        if (tap1 < 9)      a = lb[hi ? off1 : off0];        // 1 select + ds load
        else if (tap0 < 9) { _Float16 v0 = lb[off0]; a = hi ? (_Float16)0.0f : v0; }
        else               a = (_Float16)0.0f;
        av[jv] = a;
      }
      acc = __builtin_amdgcn_wmma_f32_16x16x32_f16(
          /*neg_a=*/false, av, /*neg_b=*/false, bf[t],
          /*c_mod=*/(short)0, acc, /*reuse_a=*/false, /*reuse_b=*/false);
    }

    // C fragment: lane holds N=lcol, M = r + 8*hi for acc[r];
    // M quads of 4 map to pooled pixels 2*hi and 2*hi+1.
    const float m0 = fmaxf(fmaxf(fmaxf(acc[0], acc[1]), fmaxf(acc[2], acc[3])) + bias, 0.0f);
    const float m1 = fmaxf(fmaxf(fmaxf(acc[4], acc[5]), fmaxf(acc[6], acc[7])) + bias, 0.0f);
    float* outp = f + ((b * 16 + lcol) * 28 + py) * 28 + px0 + 2 * hi;
    outp[0] = m0;
    outp[1] = m1;
  }
}

// ---------------------------------------------------------------------------
// Kernel 3: patch projection.  p[b, ph*14+pw, c*4+iy*2+ix] = f[b,c,2ph+iy,2pw+ix]
// pf[b, 784] = p @ w_patch^T + b_patch.   N=4, K=64 -> tiny, scalar.
// ---------------------------------------------------------------------------
__global__ void patch_proj_kernel(const float* __restrict__ f,
                                  const float* __restrict__ wp,
                                  const float* __restrict__ bp,
                                  float* __restrict__ pf) {
  const int t    = blockIdx.x * blockDim.x + threadIdx.x;  // < 512*196*4
  const int o    = t & 3;
  const int pidx = (t >> 2) % 196;
  const int b    = t / 784;
  const int ph   = pidx / 14;
  const int pw   = pidx % 14;

  const float* fb = f + b * 16 * 28 * 28;
  float acc = bp[o];
#pragma unroll
  for (int k = 0; k < 64; ++k) {
    const int c  = k >> 2;
    const int iy = (k >> 1) & 1;
    const int ix = k & 1;
    const float v = fb[(c * 28 + 2 * ph + iy) * 28 + (2 * pw + ix)];
    acc = fmaf(v, wp[o * 64 + k], acc);
  }
  pf[b * 784 + pidx * 4 + o] = acc;
}

// ---------------------------------------------------------------------------
// Kernel 4: fc1 GEMM via WMMA.  M=512 batch, N=64, K=784 (25 chunks of 32,
// tail zero-padded with a compile-time predicate), + bias + ReLU.
// h1: [512, 64]
// ---------------------------------------------------------------------------
__global__ void __launch_bounds__(128)
fc1_wmma_kernel(const float* __restrict__ pf,
                const float* __restrict__ w1,
                const float* __restrict__ b1,
                float* __restrict__ h1) {
  const int tid  = blockIdx.x * blockDim.x + threadIdx.x;
  const int wave = tid >> 5;               // < 32*4
  const int lane = threadIdx.x & 31;
  const int hi   = lane >> 4;
  const int lcol = lane & 15;

  const int mt  = wave >> 2;               // batch tile 0..31
  const int nt  = wave & 3;                // output tile 0..3
  const int row = mt * 16 + lcol;          // A row for this lane
  const int n   = nt * 16 + lcol;          // B/D column for this lane

  v8f acc = {};
#pragma unroll 1
  for (int t = 0; t < 25; ++t) {
    v16h av, bv;
#pragma unroll
    for (int jv = 0; jv < 16; ++jv) {
      const int k = t * 32 + ((jv >> 3) << 4) + (hi << 3) + (jv & 7);
      // 784 = 24*32 + 16: tail chunk only low K-group valid (compile-time
      // for jv; single uniform t<24 test otherwise).
      const bool valid = (jv < 8) || (t < 24);
      float a = 0.0f, bw = 0.0f;
      if (valid) {
        a  = pf[row * 784 + k];
        bw = w1[n * 784 + k];              // w_fc1 [64,784]; h1 = pf @ w^T
      }
      av[jv] = (_Float16)a;
      bv[jv] = (_Float16)bw;
    }
    acc = __builtin_amdgcn_wmma_f32_16x16x32_f16(
        false, av, false, bv, (short)0, acc, false, false);
  }

  const float bias = b1[n];
#pragma unroll
  for (int r = 0; r < 8; ++r) {
    const int mrow = mt * 16 + r + 8 * hi;       // D: M = r + 8*(lane/16)
    h1[mrow * 64 + n] = fmaxf(acc[r] + bias, 0.0f);
  }
}

// ---------------------------------------------------------------------------
// Kernel 5: fc2 (64->10) + log_softmax.  One thread per batch row.
// ---------------------------------------------------------------------------
__global__ void fc2_logsoftmax_kernel(const float* __restrict__ h1,
                                      const float* __restrict__ w2,
                                      const float* __restrict__ b2,
                                      float* __restrict__ out) {
  const int b = blockIdx.x * blockDim.x + threadIdx.x;   // < 512
  const float* hr = h1 + b * 64;
  float logits[10];
  float mx = -1e30f;
#pragma unroll
  for (int j = 0; j < 10; ++j) {
    float s = b2[j];
#pragma unroll
    for (int k = 0; k < 64; ++k) s = fmaf(hr[k], w2[j * 64 + k], s);
    logits[j] = s;
    mx = fmaxf(mx, s);
  }
  float se = 0.0f;
#pragma unroll
  for (int j = 0; j < 10; ++j) se += __expf(logits[j] - mx);
  const float lse = mx + __logf(se);
#pragma unroll
  for (int j = 0; j < 10; ++j) out[b * 10 + j] = logits[j] - lse;
}

// ---------------------------------------------------------------------------
extern "C" void kernel_launch(void* const* d_in, const int* in_sizes, int n_in,
                              void* d_out, int out_size, void* d_ws, size_t ws_size,
                              hipStream_t stream) {
  (void)in_sizes; (void)n_in; (void)out_size; (void)ws_size;

  const float* x       = (const float*)d_in[0];   // [512,1,112,112]
  const float* w_conv1 = (const float*)d_in[1];   // [8,1,3,3]
  const float* b_conv1 = (const float*)d_in[2];   // [8]
  const float* w_conv2 = (const float*)d_in[3];   // [16,8,3,3]
  const float* b_conv2 = (const float*)d_in[4];   // [16]
  const float* w_patch = (const float*)d_in[5];   // [4,64]
  const float* b_patch = (const float*)d_in[6];   // [4]
  const float* w_fc1   = (const float*)d_in[7];   // [64,784]
  const float* b_fc1   = (const float*)d_in[8];   // [64]
  const float* w_fc2   = (const float*)d_in[9];   // [10,64]
  const float* b_fc2   = (const float*)d_in[10];  // [10]
  float* out = (float*)d_out;                     // [512,10]

  // Workspace layout (floats): h | f | pf | h1   (~79 MB total)
  float* h  = (float*)d_ws;              // 512*8*56*56  = 12,845,056
  float* f  = h  + 12845056;             // 512*16*28*28 =  6,422,528
  float* pf = f  + 6422528;              // 512*784      =    401,408
  float* h1 = pf + 401408;               // 512*64       =     32,768

  // 1) conv1+pool: 512*8*56*56 threads
  conv1_pool_kernel<<<50176, 256, 0, stream>>>(x, w_conv1, b_conv1, h);

  // 2) conv2 LDS-staged im2col WMMA + fused relu/pool: 1 block per image
  conv2_wmma_pool_kernel<<<512, 128, 0, stream>>>(h, w_conv2, b_conv2, f);

  // 3) patch projection: 512*196*4 threads
  patch_proj_kernel<<<1568, 256, 0, stream>>>(f, w_patch, b_patch, pf);

  // 4) fc1 WMMA: 128 waves, 4 waves/block
  fc1_wmma_kernel<<<32, 128, 0, stream>>>(pf, w_fc1, b_fc1, h1);

  // 5) fc2 + log_softmax: one thread per batch row
  fc2_logsoftmax_kernel<<<2, 256, 0, stream>>>(h1, w_fc2, b_fc2, out);
}
